// STPatch_MGCNFormer_79182017069744
// MI455X (gfx1250) — compile-verified, matrix operand order checked by reading, not run on hardware
//
#include <hip/hip_runtime.h>
#include <hip/hip_bf16.h>

typedef unsigned short u16;
typedef __attribute__((ext_vector_type(16))) __bf16 v16bf;
typedef __attribute__((ext_vector_type(8)))  float  v8f;
typedef __attribute__((ext_vector_type(4)))  unsigned int u32x4;

// ---------------- helpers ----------------
__device__ inline u16 f2bf(float f) {
  unsigned int u = __float_as_uint(f);
  unsigned int r = u + 0x7FFFu + ((u >> 16) & 1u);
  return (u16)(r >> 16);
}
__device__ inline float bf2f(u16 h) {
  return __uint_as_float(((unsigned int)h) << 16);
}

// CDNA5 async global->LDS copy (ASYNCcnt-tracked), 16B per lane.
__device__ inline void async_load_b128(unsigned lds_addr, const void* gaddr) {
  asm volatile("global_load_async_to_lds_b128 %0, %1, off"
               :: "v"(lds_addr), "v"((unsigned long long)(size_t)gaddr)
               : "memory");
}
__device__ inline void wait_async0() {
  asm volatile("s_wait_asynccnt 0x0" ::: "memory");
}

// ---------------- generic bf16 WMMA GEMM ----------------
// C[M,N](f32/bf16) = scale*(A[M,K]bf16 @ B[K,N]bf16) (+bias) (relu) (+resid)
// Block tile 128x128x32, 256 threads = 8 wave32, each wave 32x64 (2x4 WMMA tiles).
// A and (k-contiguous) B tiles staged with global_load_async_to_lds_b128.
// Batched via blockIdx.z with two-level offsets (b1 = z/modH, b2 = z%modH).
#define GF_BIAS  1
#define GF_RELU  2
#define GF_RESID 4

__global__ __launch_bounds__(256) void gemm_bf16_kernel(
    const u16* __restrict__ A, int lda,
    const u16* __restrict__ B, int ldb, int bTrans,
    const float* __restrict__ bias,
    const float* __restrict__ resid,
    float* __restrict__ Cf, u16* __restrict__ Cb, int ldc,
    int M, int N, int K,
    long long oA1, long long oA2, long long oB1, long long oB2,
    long long oC1, long long oC2, int modH,
    float scale, int flags)
{
  __shared__ __align__(16) u16 sA[128 * 32];
  __shared__ __align__(16) u16 sB[128 * 32];   // stored transposed: [n][k]

  const int tid  = threadIdx.x;
  const int bz   = blockIdx.z;
  const long long b1 = bz / modH, b2 = bz % modH;
  const long long Ab = b1 * oA1 + b2 * oA2;
  const long long Bb = b1 * oB1 + b2 * oB2;
  const long long Cbase = b1 * oC1 + b2 * oC2;
  const int mBase = blockIdx.y * 128;
  const int nBase = blockIdx.x * 128;

  const unsigned sAbase = (unsigned)(unsigned long long)(const void*)sA;
  const unsigned sBbase = (unsigned)(unsigned long long)(const void*)sB;

  const int w = tid >> 5, lane = tid & 31;
  const int wm = w & 3, wn = w >> 2;          // 4 x 2 wave grid
  const int hv = lane >> 4, l16 = lane & 15;  // wave32 half / lane-in-16

  v8f acc[2][4];
#pragma unroll
  for (int a_ = 0; a_ < 2; a_++)
#pragma unroll
    for (int b_ = 0; b_ < 4; b_++) {
      v8f zz = {0.f,0.f,0.f,0.f,0.f,0.f,0.f,0.f};
      acc[a_][b_] = zz;
    }

  for (int k0 = 0; k0 < K; k0 += 32) {
    // ---- stage A tile (128x32): 512 chunks of 8 bf16, 2 per thread (async) ----
#pragma unroll
    for (int j = 0; j < 2; j++) {
      int c   = tid + 256 * j;
      int row = c >> 2, kc = (c & 3) * 8;
      int gr  = mBase + row, k = k0 + kc;
      if (k < K) {
        async_load_b128(sAbase + (unsigned)c * 16u,
                        A + Ab + (long long)gr * lda + k);
      } else {
        u32x4 z = {0u,0u,0u,0u};
        reinterpret_cast<u32x4*>(sA)[c] = z;
      }
    }
    // prefetch next A K-slab (global_prefetch_b8)
    if (k0 + 32 < K)
      __builtin_prefetch(A + Ab + (long long)(mBase + (tid >> 1)) * lda + k0 + 32, 0, 1);

    // ---- stage B tile transposed into sB[n][k] ----
    if (bTrans) {  // B(k,n) = Bp[n*ldb + k]  (contiguous along k -> async b128)
#pragma unroll
      for (int j = 0; j < 2; j++) {
        int c    = tid + 256 * j;
        int nIdx = c >> 2, kc = (c & 3) * 8;
        int gn = nBase + nIdx, k = k0 + kc;
        if (gn < N && k < K) {
          async_load_b128(sBbase + (unsigned)c * 16u,
                          B + Bb + (long long)gn * ldb + k);
        } else {
          u32x4 z = {0u,0u,0u,0u};
          reinterpret_cast<u32x4*>(sB)[c] = z;
        }
      }
    } else {       // B(k,n) = Bp[k*ldb + n] (activation B: gather-transpose)
#pragma unroll
      for (int j = 0; j < 2; j++) {
        int nIdx = (tid & 63) + 64 * j, kc = (tid >> 6) * 8;
        int gn = nBase + nIdx;
        union { u16 h[8]; u32x4 v; } t2;
#pragma unroll
        for (int jj = 0; jj < 8; jj++) {
          int k = k0 + kc + jj;
          t2.h[jj] = (gn < N && k < K) ? B[Bb + (long long)k * ldb + gn] : (u16)0;
        }
        reinterpret_cast<u32x4*>(sB)[nIdx * 4 + (tid >> 6)] = t2.v;
      }
    }
    wait_async0();
    __syncthreads();

    // ---- fragments per wave32 WMMA layout (A rows on lanes, B cols on lanes) ----
    union Fr { v16bf v; u32x4 u[2]; };
    Fr af[2], bfr[4];
#pragma unroll
    for (int ti = 0; ti < 2; ti++) {
      int row = wm * 32 + ti * 16 + l16;
      const u32x4* p = reinterpret_cast<const u32x4*>(sA + row * 32);
      af[ti].u[0] = p[hv];       // K = hv*8 .. +7
      af[ti].u[1] = p[2 + hv];   // K = 16 + hv*8 .. +7
    }
#pragma unroll
    for (int tj = 0; tj < 4; tj++) {
      int col = wn * 64 + tj * 16 + l16;
      const u32x4* p = reinterpret_cast<const u32x4*>(sB + col * 32);
      bfr[tj].u[0] = p[hv];
      bfr[tj].u[1] = p[2 + hv];
    }
#pragma unroll
    for (int ti = 0; ti < 2; ti++)
#pragma unroll
      for (int tj = 0; tj < 4; tj++)
        acc[ti][tj] = __builtin_amdgcn_wmma_f32_16x16x32_bf16(
            false, af[ti].v, false, bfr[tj].v, (short)0, acc[ti][tj], false, false);
    __syncthreads();
  }

  // ---- epilogue: C lane layout N=l16, M = r + 8*hv ----
#pragma unroll
  for (int ti = 0; ti < 2; ti++) {
#pragma unroll
    for (int tj = 0; tj < 4; tj++) {
      int col = nBase + wn * 64 + tj * 16 + l16;
      if (col >= N) continue;
      float bv = (flags & GF_BIAS) ? bias[col] : 0.0f;
      int rb0 = mBase + wm * 32 + ti * 16 + hv * 8;
#pragma unroll
      for (int r = 0; r < 8; r++) {
        float v = acc[ti][tj][r] * scale + bv;
        if (flags & GF_RELU) v = fmaxf(v, 0.0f);
        long long idx = Cbase + (long long)(rb0 + r) * ldc + col;
        if (flags & GF_RESID) v += resid[idx];
        if (Cf) Cf[idx] = v;
        if (Cb) Cb[idx] = f2bf(v);
      }
    }
  }
}

// ---------------- fp32 -> bf16 converters ----------------
__global__ void f2bf_kernel(const float* __restrict__ in, u16* __restrict__ out, int n) {
  int i = blockIdx.x * 256 + threadIdx.x;
  if (i < n) out[i] = f2bf(in[i]);
}
// weight (K,N) row-major -> bf16 transposed [N][K]
__global__ void f2bf_t_kernel(const float* __restrict__ in, u16* __restrict__ out,
                              int K, int N) {
  int idx = blockIdx.x * 256 + threadIdx.x;
  if (idx < K * N) {
    int k = idx / N, n = idx % N;
    out[(long long)n * K + k] = f2bf(in[idx]);
  }
}

// ---------------- RevIN + patchify ----------------
__global__ __launch_bounds__(256) void patch_kernel(
    const float* __restrict__ x, const float* __restrict__ rw, const float* __restrict__ rb,
    float* __restrict__ xpf, u16* __restrict__ xpb,
    float* __restrict__ muA, float* __restrict__ sdA)
{
  int b = blockIdx.x / 3, i = blockIdx.x % 3, tid = threadIdx.x;
  __shared__ float rs_[256], rq_[256];
  float s = 0.f, q = 0.f;
#pragma unroll
  for (int j = 0; j < 8; j++) {
    int t = tid + 256 * j;
    float v = x[((long long)b * 2048 + t) * 3 + i];
    s += v; q += v * v;
  }
  rs_[tid] = s; rq_[tid] = q;
  __syncthreads();
  for (int m = 128; m; m >>= 1) {
    if (tid < m) { rs_[tid] += rs_[tid + m]; rq_[tid] += rq_[tid + m]; }
    __syncthreads();
  }
  float mu = rs_[0] / 2048.0f;
  float var = rq_[0] / 2048.0f - mu * mu;
  float sd = sqrtf(var + 1e-5f);
  if (tid == 0) { muA[b * 3 + i] = mu; sdA[b * 3 + i] = sd; }
  float rwv = rw[i], rbv = rb[i];
#pragma unroll
  for (int j = 0; j < 8; j++) {
    int t = tid + 256 * j;
    int n = t >> 4, p = t & 15;
    float v = x[((long long)b * 2048 + t) * 3 + i];
    float xn = (v - mu) / sd * rwv + rbv;
    long long o = ((long long)(b * 128 + n) * 3 + i) * 16 + p;
    xpf[o] = xn; xpb[o] = f2bf(xn);
  }
}

// ---------------- e(b,n,i,d) + pos(n,d) -> x0(b,i,n,d) ----------------
__global__ void embed_rearrange_kernel(const float* __restrict__ e, const float* __restrict__ pos,
                                       float* __restrict__ xf, u16* __restrict__ xb)
{
  long long gid = (long long)blockIdx.x * 256 + threadIdx.x;
  int d = gid & 511;
  long long r = gid >> 9;
  int i = (int)(r % 3), n = (int)((r / 3) % 128), b = (int)(r / 384);
  float v = e[gid] + pos[n * 512 + d];
  long long o = ((long long)(b * 3 + i) * 128 + n) * 512 + d;
  xf[o] = v; xb[o] = f2bf(v);
}

// ---------------- eout(b,i,n,d) -> z(i,n,b,d) ----------------
__global__ void z_rearrange_kernel(const float* __restrict__ in,
                                   float* __restrict__ zf, u16* __restrict__ zb)
{
  long long gid = (long long)blockIdx.x * 256 + threadIdx.x;
  int d = gid & 511;
  long long r = gid >> 9;
  int n = (int)(r % 128), i = (int)((r / 128) % 3), b = (int)(r / 384);
  float v = in[gid];
  long long o = ((long long)(i * 128 + n) * 128 + b) * 512 + d;
  zf[o] = v; zb[o] = f2bf(v);
}

// ---------------- LayerNorm(ra + gb) over D=512, wave per row ----------------
__global__ __launch_bounds__(256) void ln512_kernel(
    const float* __restrict__ ra, const float* __restrict__ gb,
    const float* __restrict__ w, const float* __restrict__ bia,
    float* __restrict__ outf, u16* __restrict__ outb)
{
  int row = blockIdx.x * 8 + (threadIdx.x >> 5);
  int lane = threadIdx.x & 31;
  long long base = (long long)row * 512;
  float v[16]; float s = 0.f;
#pragma unroll
  for (int j = 0; j < 16; j++) {
    int d = lane + 32 * j;
    v[j] = ra[base + d] + gb[base + d];
    s += v[j];
  }
  for (int m = 16; m; m >>= 1) s += __shfl_xor(s, m);
  float mu = s / 512.0f;
  float ss = 0.f;
#pragma unroll
  for (int j = 0; j < 16; j++) { float t = v[j] - mu; ss += t * t; }
  for (int m = 16; m; m >>= 1) ss += __shfl_xor(ss, m);
  float inv = rsqrtf(ss / 512.0f + 1e-5f);
#pragma unroll
  for (int j = 0; j < 16; j++) {
    int d = lane + 32 * j;
    float o = (v[j] - mu) * inv * w[d] + bia[d];
    if (outf) outf[base + d] = o;
    if (outb) outb[base + d] = f2bf(o);
  }
}

// ---------------- softmax over rows of 128 (bf16 in place), wave per row ----------------
__global__ __launch_bounds__(256) void softmax128_kernel(u16* __restrict__ sc)
{
  long long row = (long long)blockIdx.x * 8 + (threadIdx.x >> 5);
  int lane = threadIdx.x & 31;
  long long base = row * 128;
  float v[4];
#pragma unroll
  for (int j = 0; j < 4; j++) v[j] = bf2f(sc[base + lane + 32 * j]);
  float mx = fmaxf(fmaxf(v[0], v[1]), fmaxf(v[2], v[3]));
  for (int m = 16; m; m >>= 1) mx = fmaxf(mx, __shfl_xor(mx, m));
  float s = 0.f;
#pragma unroll
  for (int j = 0; j < 4; j++) { v[j] = expf(v[j] - mx); s += v[j]; }
  for (int m = 16; m; m >>= 1) s += __shfl_xor(s, m);
  float inv = 1.0f / s;
#pragma unroll
  for (int j = 0; j < 4; j++) sc[base + lane + 32 * j] = f2bf(v[j] * inv);
}

// ---------------- Abar = mean_m softmax(dist[m], -1), bf16 out ----------------
__global__ __launch_bounds__(128) void abar_kernel(const float* __restrict__ dist,
                                                   u16* __restrict__ ab)
{
  int u_ = blockIdx.x, v_ = threadIdx.x;
  __shared__ float sh[128];
  float acc = 0.f;
  for (int m = 0; m < 3; m++) {
    float x = dist[((long long)m * 128 + u_) * 128 + v_];
    sh[v_] = x; __syncthreads();
    for (int s = 64; s; s >>= 1) { if (v_ < s) sh[v_] = fmaxf(sh[v_], sh[v_ + s]); __syncthreads(); }
    float mx = sh[0]; __syncthreads();
    float e = expf(x - mx);
    sh[v_] = e; __syncthreads();
    for (int s = 64; s; s >>= 1) { if (v_ < s) sh[v_] += sh[v_ + s]; __syncthreads(); }
    float sum = sh[0]; __syncthreads();
    acc += e / sum;
  }
  ab[u_ * 128 + v_] = f2bf(acc / 3.0f);
}

// ---------------- gate + combine + denorm -> out1 ----------------
__global__ void gate_kernel(const float* __restrict__ dy, const float* __restrict__ st,
                            const float* __restrict__ rw, const float* __restrict__ rb,
                            const float* __restrict__ muA, const float* __restrict__ sdA,
                            float* __restrict__ out1)
{
  long long gid = (long long)blockIdx.x * 256 + threadIdx.x; // 786432 total
  int p = gid & 15;
  long long row = gid >> 4;            // row = (i*128+n)*128 + b
  int b = (int)(row & 127);
  long long r = row >> 7;
  int n = (int)(r & 127), i = (int)(r >> 7);
  float d = dy[gid], s = st[gid];
  float g = 1.0f / (1.0f + expf(-(d + s)));
  float zc = g * d + (1.0f - g) * s;
  float val = (zc - rb[i]) / rw[i] * sdA[b * 3 + i] + muA[b * 3 + i];
  out1[(((long long)(b * 128 + n)) * 16 + p) * 3 + i] = val;
}

// ---------------- time_scores (out2) + region_flow ----------------
__global__ __launch_bounds__(128) void ts_kernel(const float* __restrict__ xp,
                                                 float* __restrict__ out2,
                                                 float* __restrict__ rf)
{
  int b = blockIdx.x / 3, i = blockIdx.x % 3, tid = threadIdx.x;
  __shared__ float mf[16];
  if (tid < 16) {
    float m = -1e30f, s = 0.f;
    for (int n = 0; n < 128; n++) {
      float v = xp[((long long)(b * 128 + n) * 3 + i) * 16 + tid];
      m = fmaxf(m, v); s += v;
    }
    mf[tid] = m;
    rf[((long long)(b * 3 + i)) * 16 + tid] = s / 128.0f;
  }
  __syncthreads();
  {
    int n = tid;
    float s = 0.f;
    for (int p = 0; p < 16; p++)
      s += xp[((long long)(b * 128 + n) * 3 + i) * 16 + p] / (mf[p] + 1e-6f);
    float t = s / 16.0f + 1e-6f;
    out2[((long long)(b * 128 + n)) * 3 + i] = fminf(fmaxf(t, 1e-6f), 1e6f);
  }
}

// ---------------- pairwise region distances + surrogate top-k -> out3 ----------------
__global__ __launch_bounds__(128) void pmrs_kernel(const float* __restrict__ rf,
                                                   float* __restrict__ out3)
{
  int u_ = blockIdx.x, v_ = threadIdx.x;
  __shared__ float pm[128];
  float m = 0.f;
  for (int p = 0; p < 16; p++) {
    float s = 0.f;
    for (int i = 0; i < 3; i++) {
      float d = rf[((long long)(u_ * 3 + i)) * 16 + p] - rf[((long long)(v_ * 3 + i)) * 16 + p];
      s += d * d;
    }
    m = fmaxf(m, sqrtf(s));
  }
  pm[v_] = m;
  __syncthreads();
  if (v_ == 0) {
    unsigned int seed = (unsigned int)u_ * 2654435761u + 12345u;
    float mx = -1e30f;
    for (int j = 0; j < 12; j++) {           // kk = max(1, int(0.1*(B-1))) = 12
      seed = seed * 1664525u + 1013904223u;
      int vv = (int)((seed >> 8) % 127u);
      if (vv >= u_) vv++;
      mx = fmaxf(mx, pm[vv]);
    }
    out3[u_] = fminf(fmaxf(mx + 1e-6f, 1e-6f), 1e6f);
  }
}

// ---------------- host-side launch ----------------
static inline void launch_gemm(hipStream_t s, const u16* A, int lda, const u16* B, int ldb, int bT,
                               const float* bias, const float* resid,
                               float* Cf, u16* Cb, int ldc,
                               int M, int N, int K,
                               long long oA1, long long oA2, long long oB1, long long oB2,
                               long long oC1, long long oC2, int modH, int nb,
                               float scale, int flags)
{
  dim3 g((N + 127) / 128, M / 128, nb);
  gemm_bf16_kernel<<<g, 256, 0, s>>>(A, lda, B, ldb, bT, bias, resid, Cf, Cb, ldc,
                                     M, N, K, oA1, oA2, oB1, oB2, oC1, oC2, modH, scale, flags);
}

extern "C" void kernel_launch(void* const* d_in, const int* in_sizes, int n_in,
                              void* d_out, int out_size, void* d_ws, size_t ws_size,
                              hipStream_t stream) {
  (void)in_sizes; (void)out_size; (void)ws_size;

  constexpr int NROW = 49152;                 // 384 * 128 token rows
  constexpr long long NE = (long long)NROW * 512;

  // ---- input map: x, dist_mats, then params flattened (jax sorted-key order) ----
  const float* P[64];
  for (int t = 0; t < 53 && t < n_in; t++) P[t] = (const float*)d_in[t];
  const float* x_in   = P[0];   // (128,2048,3)
  const float* dist   = P[1];   // (3,128,128)
  const float* Wp_b   = P[2];   const float* Wp_w   = P[3];    // (16,512)
  const float* ef1_b  = P[4];   const float* ef1_w  = P[5];    // (512,256)
  const float* ef2_b  = P[6];   const float* ef2_w  = P[7];    // (256,512)
  const float* ek_b   = P[8];   const float* ek_w   = P[9];
  const float* eln1b  = P[10];  const float* eln1w  = P[11];
  const float* eln2b  = P[12];  const float* eln2w  = P[13];
  const float* eo_b   = P[14];  const float* eo_w   = P[15];
  const float* eq_b   = P[16];  const float* eq_w   = P[17];
  const float* ev_b   = P[18];  const float* ev_w   = P[19];
  const float* g0_b   = P[20];  const float* g0_w   = P[21];
  const float* g1_b   = P[22];  const float* g1_w   = P[23];
  const float* g2_b   = P[24];  const float* g2_w   = P[25];
  const float* pos    = P[26];  // (128,512)
  const float* d1_b   = P[27];  const float* d1_w   = P[28];   // (512,256)
  const float* d2_b   = P[29];  const float* d2_w   = P[30];   // (256,16)
  const float* s1_b   = P[31];  const float* s1_w   = P[32];
  const float* s2_b   = P[33];  const float* s2_w   = P[34];
  const float* revb   = P[35];  const float* revw   = P[36];
  const float* sf1_b  = P[37];  const float* sf1_w  = P[38];   // (512,512)
  const float* sf2_b  = P[39];  const float* sf2_w  = P[40];   // (512,512)
  const float* sk_b   = P[41];  const float* sk_w   = P[42];
  const float* sln1b  = P[43];  const float* sln1w  = P[44];
  const float* sln2b  = P[45];  const float* sln2w  = P[46];
  const float* so_b   = P[47];  const float* so_w   = P[48];
  const float* sq_b   = P[49];  const float* sq_w   = P[50];
  const float* sv_b   = P[51];  const float* sv_w   = P[52];

  float* out1 = (float*)d_out;          // z_den (128,128,16,3)
  float* out2 = out1 + 786432;          // ts    (128,128,3,1)
  float* out3 = out2 + 49152;           // rs    (128,1,1)

  // ---- workspace bump allocator ----
  char* wsb = (char*)d_ws;
  size_t cur = 0;
  auto alloc = [&](size_t bytes) -> void* {
    void* p = wsb + cur;
    cur += (bytes + 255) & ~(size_t)255;
    return p;
  };

  float* XPF = (float*)alloc(786432 * 4);          // xp f32 (b,n,i,p)
  u16*   XPB = (u16*)  alloc(786432 * 2);
  float* MU  = (float*)alloc(384 * 4);
  float* SD  = (float*)alloc(384 * 4);
  float* RF  = (float*)alloc(128 * 3 * 16 * 4);
  float* FT  = (float*)alloc(NE * 4);              // generic f32 gemm out
  float* FR  = (float*)alloc(NE * 4);              // running residual
  float* FZ  = (float*)alloc(NE * 4);              // z f32
  float* FH1 = (float*)alloc(NE * 4);
  float* FH2 = (float*)alloc(NE * 4);
  u16*   BX  = (u16*)alloc(NE * 2);                // current activation bf16
  u16*   BQ  = (u16*)alloc(NE * 2);
  u16*   BK  = (u16*)alloc(NE * 2);
  u16*   BV  = (u16*)alloc(NE * 2);
  u16*   BO  = (u16*)alloc(NE * 2);
  u16*   BT  = (u16*)alloc(NE * 2);
  u16*   BH1 = (u16*)alloc(NE * 2);
  u16*   BH2 = (u16*)alloc(NE * 2);
  u16*   BZ  = (u16*)alloc(NE * 2);
  u16*   BDY = (u16*)alloc(NE * 2);
  u16*   BF1 = (u16*)alloc(NE * 2);                // ffn hidden (up to 512 wide)
  u16*   BS  = (u16*)alloc((size_t)3072 * 16384 * 2); // attention scores bf16
  float* FDO = (float*)alloc(786432 * 4);
  float* FSO = (float*)alloc(786432 * 4);
  u16*   ABB = (u16*)alloc(16384 * 2);

  // weights: convert + transpose to [N][K] bf16 so GEMM-B is k-contiguous
  auto convT = [&](const float* src, int K, int N) -> u16* {
    u16* dst = (u16*)alloc((size_t)K * N * 2);
    f2bf_t_kernel<<<(K * N + 255) / 256, 256, 0, stream>>>(src, dst, K, N);
    return dst;
  };
  u16* WpT  = convT(Wp_w, 16, 512);
  u16* EQT  = convT(eq_w, 512, 512);  u16* EKT = convT(ek_w, 512, 512);
  u16* EVT  = convT(ev_w, 512, 512);  u16* EOT = convT(eo_w, 512, 512);
  u16* EF1T = convT(ef1_w, 512, 256); u16* EF2T = convT(ef2_w, 256, 512);
  u16* SQT  = convT(sq_w, 512, 512);  u16* SKT = convT(sk_w, 512, 512);
  u16* SVT  = convT(sv_w, 512, 512);  u16* SOT = convT(so_w, 512, 512);
  u16* SF1T = convT(sf1_w, 512, 512); u16* SF2T = convT(sf2_w, 512, 512);
  u16* GWT[3] = { convT(g0_w, 512, 512), convT(g1_w, 512, 512), convT(g2_w, 512, 512) };
  const float* GB[3] = { g0_b, g1_b, g2_b };
  u16* D1T = convT(d1_w, 512, 256);   u16* D2T = convT(d2_w, 256, 16);
  u16* S1T = convT(s1_w, 512, 256);   u16* S2T = convT(s2_w, 256, 16);

  // ---- 1) RevIN + patchify ----
  patch_kernel<<<384, 256, 0, stream>>>(x_in, revw, revb, XPF, XPB, MU, SD);
  // ---- 2) Abar ----
  abar_kernel<<<128, 128, 0, stream>>>(dist, ABB);

  // ---- 3) patch embed: (49152x16) @ (16x512) ----
  launch_gemm(stream, XPB, 16, WpT, 16, 1, Wp_b, nullptr, FT, nullptr, 512,
              NROW, 512, 16, 0,0,0,0,0,0, 1, 1, 1.0f, GF_BIAS);
  embed_rearrange_kernel<<<(int)(NE / 256), 256, 0, stream>>>(FT, pos, FR, BX);

  // ---- transformer block (shared shape for enc and sp) ----
  auto attn_block = [&](const u16* qw, const float* qb, const u16* kw, const float* kb,
                        const u16* vw, const float* vb, const u16* ow, const float* ob,
                        const float* l1w, const float* l1b, const float* l2w, const float* l2b,
                        const u16* f1w, const float* f1b, int dff,
                        const u16* f2w, const float* f2b,
                        const float* residIn, float* ln1f, u16* ln1bf,
                        float* ln2f, u16* ln2bf) {
    launch_gemm(stream, BX, 512, qw, 512, 1, qb, nullptr, nullptr, BQ, 512,
                NROW, 512, 512, 0,0,0,0,0,0, 1, 1, 1.0f, GF_BIAS);
    launch_gemm(stream, BX, 512, kw, 512, 1, kb, nullptr, nullptr, BK, 512,
                NROW, 512, 512, 0,0,0,0,0,0, 1, 1, 1.0f, GF_BIAS);
    launch_gemm(stream, BX, 512, vw, 512, 1, vb, nullptr, nullptr, BV, 512,
                NROW, 512, 512, 0,0,0,0,0,0, 1, 1, 1.0f, GF_BIAS);
    // scores = (q @ k^T) / 8 over (s,h): batch 3072
    launch_gemm(stream, BQ, 512, BK, 512, 1, nullptr, nullptr, nullptr, BS, 128,
                128, 128, 64, 65536, 64, 65536, 64, 131072, 16384, 8, 3072,
                0.125f, 0);
    softmax128_kernel<<<49152, 256, 0, stream>>>(BS);
    // o = attn @ v
    launch_gemm(stream, BS, 128, BV, 512, 0, nullptr, nullptr, nullptr, BO, 512,
                128, 64, 128, 131072, 16384, 65536, 64, 65536, 64, 8, 3072,
                1.0f, 0);
    // output projection + LN1
    launch_gemm(stream, BO, 512, ow, 512, 1, ob, nullptr, FT, nullptr, 512,
                NROW, 512, 512, 0,0,0,0,0,0, 1, 1, 1.0f, GF_BIAS);
    ln512_kernel<<<6144, 256, 0, stream>>>(residIn, FT, l1w, l1b, ln1f, ln1bf);
    // FFN
    launch_gemm(stream, ln1bf, 512, f1w, 512, 1, f1b, nullptr, nullptr, BF1, dff,
                NROW, dff, 512, 0,0,0,0,0,0, 1, 1, 1.0f, GF_BIAS | GF_RELU);
    launch_gemm(stream, BF1, dff, f2w, dff, 1, f2b, nullptr, FT, nullptr, 512,
                NROW, 512, dff, 0,0,0,0,0,0, 1, 1, 1.0f, GF_BIAS);
    ln512_kernel<<<6144, 256, 0, stream>>>(ln1f, FT, l2w, l2b, ln2f, ln2bf);
  };

  // ---- 4) encoder block on x0 (FR f32, BX bf16) -> eout in FR/BX ----
  attn_block(EQT, eq_b, EKT, ek_b, EVT, ev_b, EOT, eo_b,
             eln1w, eln1b, eln2w, eln2b, EF1T, ef1_b, 256, EF2T, ef2_b,
             FR, FR, BX, FR, BX);

  // ---- 5) rearrange to z (i,n,b,d) ----
  z_rearrange_kernel<<<(int)(NE / 256), 256, 0, stream>>>(FR, FZ, BZ);
  hipMemcpyAsync(BX, BZ, NE * 2, hipMemcpyDeviceToDevice, stream);

  // ---- 6) spatial block on z -> dy_z (bf16 in BDY) ----
  attn_block(SQT, sq_b, SKT, sk_b, SVT, sv_b, SOT, so_b,
             sln1w, sln1b, sln2w, sln2b, SF1T, sf1_b, 512, SF2T, sf2_b,
             FZ, FR, BX, nullptr, BDY);

  // ---- 7) GCN: h0 = z; h = relu(Abar @ h @ W + b) + h, 3x ----
  const float* hF = FZ; const u16* hB = BZ;
  float* hOutF[3] = { FH1, FH2, FH1 };
  u16*   hOutB[3] = { BH1, BH2, BH1 };
  for (int it = 0; it < 3; it++) {
    // t = Abar @ h  (batched over 384 slices)
    launch_gemm(stream, ABB, 128, hB, 512, 0, nullptr, nullptr, nullptr, BT, 512,
                128, 512, 128, 0, 0, 65536, 0, 65536, 0, 1, 384, 1.0f, 0);
    // h' = relu(t @ W + b) + h
    launch_gemm(stream, BT, 512, GWT[it], 512, 1, GB[it], hF, hOutF[it], hOutB[it], 512,
                NROW, 512, 512, 0,0,0,0,0,0, 1, 1, 1.0f,
                GF_BIAS | GF_RELU | GF_RESID);
    hF = hOutF[it]; hB = hOutB[it];
  }

  // ---- 8) projection heads ----
  launch_gemm(stream, BDY, 512, D1T, 512, 1, d1_b, nullptr, nullptr, BF1, 256,
              NROW, 256, 512, 0,0,0,0,0,0, 1, 1, 1.0f, GF_BIAS | GF_RELU);
  launch_gemm(stream, BF1, 256, D2T, 256, 1, d2_b, nullptr, FDO, nullptr, 16,
              NROW, 16, 256, 0,0,0,0,0,0, 1, 1, 1.0f, GF_BIAS);
  launch_gemm(stream, hB, 512, S1T, 512, 1, s1_b, nullptr, nullptr, BF1, 256,
              NROW, 256, 512, 0,0,0,0,0,0, 1, 1, 1.0f, GF_BIAS | GF_RELU);
  launch_gemm(stream, BF1, 256, S2T, 256, 1, s2_b, nullptr, FSO, nullptr, 16,
              NROW, 16, 256, 0,0,0,0,0,0, 1, 1, 1.0f, GF_BIAS);

  // ---- 9) gate + denorm -> out1 ----
  gate_kernel<<<3072, 256, 0, stream>>>(FDO, FSO, revw, revb, MU, SD, out1);

  // ---- 10) statistics -> out2, out3 ----
  ts_kernel<<<384, 128, 0, stream>>>(XPF, out2, RF);
  pmrs_kernel<<<128, 128, 0, stream>>>(RF, out3);
}
